// SC_TGN_20409684590939
// MI455X (gfx1250) — matrix operand order repaired and checked by reading, not compiled, not run on hardware
//
#include <hip/hip_runtime.h>
#include <hip/hip_bf16.h>

typedef __attribute__((ext_vector_type(16))) _Float16 v16h;
typedef __attribute__((ext_vector_type(8)))  _Float16 v8h;
typedef __attribute__((ext_vector_type(8)))  float    v8f;

// ---------------------------------------------------------------------------
// WMMA helpers (CDNA5 / gfx1250, wave32).
//
// f16 A-matrix 16x32 fragment (ISA 7.12.2): lane l (l<16) holds row M=l,
// K pairs {0..7} in VGPR0-3 and {16..23} in VGPR4-7; lanes 16-31 hold row
// M=l-16, K {8..15} and {24..31}.  So with ko = (l>=16)?8:0, the fragment is
// two contiguous 8-half chunks at [k0+ko, +8) and [k0+16+ko, +8).
//
// f16 B-matrix 32x16 fragment: lane l (l<16) holds column N=l, K {0..7} and {16..23}
// (ascending pairs per VGPR); lanes 16-31 hold column N=l-16, K {8..15} and {24..31}.
// We pre-pack weights so lane l reads 16 contiguous halves at frag*512+l*16.
//
// f32 C/D 16x16: VGPR j, lane l -> row M = j + 8*(l>=16), col N = l&15.
// ---------------------------------------------------------------------------

__device__ __forceinline__ v8f wmma16(v16h a, v16h b, v8f c) {
  return __builtin_amdgcn_wmma_f32_16x16x32_f16(
      /*neg_a=*/false, a, /*neg_b=*/false, b,
      /*c_mod=*/(short)0, c, /*reuse_a=*/false, /*reuse_b=*/false);
}

__device__ __forceinline__ v16h load_a_frag(const _Float16* row, int k /* = k0 + ko */) {
  union { v16h v; v8h h[2]; } u;
  u.h[0] = *(const v8h*)(row + k);
  u.h[1] = *(const v8h*)(row + k + 16);
  return u.v;
}

__device__ __forceinline__ v16h load_b_frag(const _Float16* pw, int frag, int lane) {
  const _Float16* p = pw + (size_t)frag * 512 + (size_t)lane * 16;
  union { v16h v; v8h h[2]; } u;
  u.h[0] = *(const v8h*)p;
  u.h[1] = *(const v8h*)(p + 8);
  return u.v;
}

// ---------------------------------------------------------------------------
// Pack an f32 weight matrix (KD x ND, or its transpose) into f16 WMMA
// B-fragments.  Fragment id = nt*KT + kt (kt contiguous for streaming).
// Half index inside buffer: frag*512 + lane*16 + e, where
//   k = kt*32 + 16*(e>=8) + 8*(lane>=16) + (e&7),  n = nt*16 + (lane&15).
// ---------------------------------------------------------------------------
__global__ void pack_wmma_b(const float* __restrict__ src, _Float16* __restrict__ dst,
                            int KD, int ND, int KT, int transposed) {
  int idx = blockIdx.x * blockDim.x + threadIdx.x;
  int total = KD * ND;
  if (idx >= total) return;
  int frag = idx >> 9;
  int lane = (idx >> 4) & 31;
  int e    = idx & 15;
  int nt = frag / KT;
  int kt = frag - nt * KT;
  int k = kt * 32 + ((e >> 3) << 4) + ((lane >> 4) << 3) + (e & 7);
  int n = nt * 16 + (lane & 15);
  float v = transposed ? src[(size_t)n * KD + k] : src[(size_t)k * ND + n];
  dst[idx] = (_Float16)v;
}

// ---------------------------------------------------------------------------
// Kernel 1: per-edge message MLP + scatter-add aggregation.
// Block = 16 edges, 256 threads (8 waves); wave w owns N-tile w (16 cols).
// x[16][320] = [src_mem | dst_mem | edge_feat | cos(t*w+b)] staged as f16.
// ---------------------------------------------------------------------------
__global__ __launch_bounds__(256) void tgn_edge_msg(
    const int* __restrict__ src_nodes, const int* __restrict__ dst_nodes,
    const float* __restrict__ edge_feat, const float* __restrict__ timestamps,
    const float* __restrict__ memory,
    const float* __restrict__ time_w, const float* __restrict__ time_b,
    const _Float16* __restrict__ pW1, const float* __restrict__ msg_b1,
    const _Float16* __restrict__ pW2, const float* __restrict__ msg_b2,
    float* __restrict__ aggr, float* __restrict__ counts, int E) {
  __shared__ __align__(16) _Float16 xs[16][320];
  __shared__ __align__(16) _Float16 hs[16][128];
  __shared__ int s_src[16], s_dst[16];

  const int tid  = threadIdx.x;
  const int lane = tid & 31;
  const int wave = tid >> 5;
  const int eb   = blockIdx.x * 16;

  if (tid < 16) {
    int e = min(eb + tid, E - 1);
    s_src[tid] = src_nodes[e];
    s_dst[tid] = dst_nodes[e];
  }
  // gather src/dst memory rows (f32 -> f16)
  for (int i = tid; i < 16 * 128; i += 256) {
    int r = i >> 7, k = i & 127;
    int e = min(eb + r, E - 1);
    int si = src_nodes[e], di = dst_nodes[e];
    xs[r][k]       = (_Float16)memory[(size_t)si * 128 + k];
    xs[r][128 + k] = (_Float16)memory[(size_t)di * 128 + k];
  }
  // edge features + time encoding
  for (int i = tid; i < 16 * 32; i += 256) {
    int r = i >> 5, c = i & 31;
    int e = min(eb + r, E - 1);
    xs[r][256 + c] = (_Float16)edge_feat[(size_t)e * 32 + c];
    float t = timestamps[e];
    xs[r][288 + c] = (_Float16)cosf(t * time_w[c] + time_b[c]);
  }
  __syncthreads();

  const int r     = lane & 15;
  const int ko    = (lane >> 4) << 3;  // 0 or 8
  const int nl    = lane & 15;
  const int rbase = (lane >> 4) << 3;  // 0 or 8
  const int nt    = wave;

  // layer 1: [16,320] @ [320,128] -> relu -> hs
  v8f acc = {};
  for (int kt = 0; kt < 10; ++kt) {
    if (kt + 1 < 10)
      __builtin_prefetch(pW1 + ((size_t)(nt * 10 + kt + 1)) * 512 + (size_t)lane * 16, 0, 3);
    v16h a = load_a_frag(&xs[r][0], kt * 32 + ko);
    v16h b = load_b_frag(pW1, nt * 10 + kt, lane);
    acc = wmma16(a, b, acc);
  }
  {
    int n = nt * 16 + nl;
    float bb = msg_b1[n];
#pragma unroll
    for (int j = 0; j < 8; ++j)
      hs[rbase + j][n] = (_Float16)fmaxf(acc[j] + bb, 0.0f);
  }
  __syncthreads();

  // layer 2: [16,128] @ [128,128] -> msg, then scatter-add to both endpoints
  v8f acc2 = {};
  for (int kt = 0; kt < 4; ++kt) {
    v16h a = load_a_frag(&hs[r][0], kt * 32 + ko);
    v16h b = load_b_frag(pW2, nt * 4 + kt, lane);
    acc2 = wmma16(a, b, acc2);
  }
  {
    int n = nt * 16 + nl;
    float bb = msg_b2[n];
#pragma unroll
    for (int j = 0; j < 8; ++j) {
      int rj = rbase + j;
      if (eb + rj < E) {
        float v = acc2[j] + bb;
        atomicAdd(&aggr[(size_t)s_src[rj] * 128 + n], v);
        atomicAdd(&aggr[(size_t)s_dst[rj] * 128 + n], v);
      }
    }
  }
  if (tid < 16 && eb + tid < E) {
    atomicAdd(&counts[s_src[tid]], 1.0f);
    atomicAdd(&counts[s_dst[tid]], 1.0f);
  }
}

// ---------------------------------------------------------------------------
// Kernel 2: GRU memory update over all nodes.
// Block = 16 nodes; gi = mean @ W_ih.T (128->384), gh = mem @ W_hh.T.
// 48 WMMA tasks split across 8 waves, then elementwise gates.
// ---------------------------------------------------------------------------
__global__ __launch_bounds__(256) void tgn_gru(
    const float* __restrict__ memory, const float* __restrict__ aggr,
    const float* __restrict__ counts,
    const _Float16* __restrict__ pGIH, const _Float16* __restrict__ pGHH,
    const float* __restrict__ b_ih, const float* __restrict__ b_hh,
    float* __restrict__ newmem, int N) {
  __shared__ __align__(16) _Float16 smean[16][128];
  __shared__ __align__(16) _Float16 smem[16][128];
  __shared__ float sgi[16][384];
  __shared__ float sgh[16][384];

  const int tid  = threadIdx.x;
  const int lane = tid & 31;
  const int wave = tid >> 5;
  const int nb   = blockIdx.x * 16;

  for (int i = tid; i < 2048; i += 256) {
    int rr = i >> 7, k = i & 127;
    int node = min(nb + rr, N - 1);
    float c  = counts[node];
    float mv = aggr[(size_t)node * 128 + k] / fmaxf(c, 1e-9f);
    smean[rr][k] = (_Float16)mv;
    smem[rr][k]  = (_Float16)memory[(size_t)node * 128 + k];
  }
  __syncthreads();

  const int r     = lane & 15;
  const int ko    = (lane >> 4) << 3;
  const int nl    = lane & 15;
  const int rbase = (lane >> 4) << 3;

  for (int task = wave; task < 48; task += 8) {
    bool hh = task >= 24;
    int  nt = hh ? task - 24 : task;
    const _Float16* pw = hh ? pGHH : pGIH;
    _Float16 (*A)[128] = hh ? smem : smean;
    v8f acc = {};
    for (int kt = 0; kt < 4; ++kt) {
      v16h a = load_a_frag(&A[r][0], kt * 32 + ko);
      v16h b = load_b_frag(pw, nt * 4 + kt, lane);
      acc = wmma16(a, b, acc);
    }
    int n = nt * 16 + nl;
    float bias = (hh ? b_hh : b_ih)[n];
    float (*G)[384] = hh ? sgh : sgi;
#pragma unroll
    for (int j = 0; j < 8; ++j) G[rbase + j][n] = acc[j] + bias;
  }
  __syncthreads();

  for (int i = tid; i < 2048; i += 256) {
    int rr = i >> 7, m = i & 127;
    int node = nb + rr;
    if (node >= N) continue;
    float ir = sgi[rr][m], iz = sgi[rr][128 + m], inn = sgi[rr][256 + m];
    float hr = sgh[rr][m], hz = sgh[rr][128 + m], hn  = sgh[rr][256 + m];
    float mem = memory[(size_t)node * 128 + m];
    float rg  = 1.0f / (1.0f + expf(-(ir + hr)));
    float z   = 1.0f / (1.0f + expf(-(iz + hz)));
    float ng  = tanhf(inn + rg * hn);
    float hnew = (1.0f - z) * ng + z * mem;
    newmem[(size_t)node * 128 + m] = (counts[node] > 0.0f) ? hnew : mem;
  }
}

// ---------------------------------------------------------------------------
// Kernel 3: embedding MLP for both endpoints + dot product.
// Block = 16 edges; 8 waves = {src,dst} x 4 N-tiles of 16.
// ---------------------------------------------------------------------------
__global__ __launch_bounds__(256) void tgn_embed(
    const int* __restrict__ src_nodes, const int* __restrict__ dst_nodes,
    const float* __restrict__ src_features, const float* __restrict__ dst_features,
    const float* __restrict__ newmem,
    const _Float16* __restrict__ pEW1, const float* __restrict__ eb1,
    const _Float16* __restrict__ pEW2, const float* __restrict__ eb2,
    float* __restrict__ out, int E) {
  __shared__ __align__(16) _Float16 cs[2][16][192];
  __shared__ __align__(16) _Float16 hsd[2][16][64];
  __shared__ float esd[2][16][64];

  const int tid   = threadIdx.x;
  const int lane  = tid & 31;
  const int wave  = tid >> 5;
  const int ebase = blockIdx.x * 16;

  for (int i = tid; i < 2048; i += 256) {
    int rr = i >> 7, k = i & 127;
    int e = min(ebase + rr, E - 1);
    cs[0][rr][k] = (_Float16)newmem[(size_t)src_nodes[e] * 128 + k];
    cs[1][rr][k] = (_Float16)newmem[(size_t)dst_nodes[e] * 128 + k];
  }
  for (int i = tid; i < 1024; i += 256) {
    int rr = i >> 6, c = i & 63;
    int e = min(ebase + rr, E - 1);
    cs[0][rr][128 + c] = (_Float16)src_features[(size_t)e * 64 + c];
    cs[1][rr][128 + c] = (_Float16)dst_features[(size_t)e * 64 + c];
  }
  __syncthreads();

  const int r     = lane & 15;
  const int ko    = (lane >> 4) << 3;
  const int nl    = lane & 15;
  const int rbase = (lane >> 4) << 3;
  const int side  = wave >> 2;
  const int nt    = wave & 3;

  // layer 1: [16,192] @ [192,64], relu
  {
    v8f acc = {};
    for (int kt = 0; kt < 6; ++kt) {
      v16h a = load_a_frag(&cs[side][r][0], kt * 32 + ko);
      v16h b = load_b_frag(pEW1, nt * 6 + kt, lane);
      acc = wmma16(a, b, acc);
    }
    int n = nt * 16 + nl;
    float bb = eb1[n];
#pragma unroll
    for (int j = 0; j < 8; ++j)
      hsd[side][rbase + j][n] = (_Float16)fmaxf(acc[j] + bb, 0.0f);
  }
  __syncthreads();

  // layer 2: [16,64] @ [64,64]
  {
    v8f acc = {};
    for (int kt = 0; kt < 2; ++kt) {
      v16h a = load_a_frag(&hsd[side][r][0], kt * 32 + ko);
      v16h b = load_b_frag(pEW2, nt * 2 + kt, lane);
      acc = wmma16(a, b, acc);
    }
    int n = nt * 16 + nl;
    float bb = eb2[n];
#pragma unroll
    for (int j = 0; j < 8; ++j) esd[side][rbase + j][n] = acc[j] + bb;
  }
  __syncthreads();

  if (tid < 16 && ebase + tid < E) {
    float s = 0.0f;
#pragma unroll 8
    for (int c = 0; c < 64; ++c) s += esd[0][tid][c] * esd[1][tid][c];
    out[ebase + tid] = s;
  }
}

// ---------------------------------------------------------------------------
extern "C" void kernel_launch(void* const* d_in, const int* in_sizes, int n_in,
                              void* d_out, int out_size, void* d_ws, size_t ws_size,
                              hipStream_t stream) {
  const int*   src_nodes  = (const int*)d_in[0];
  const int*   dst_nodes  = (const int*)d_in[1];
  const float* edge_feat  = (const float*)d_in[2];
  const float* timestamps = (const float*)d_in[3];
  const float* src_feat   = (const float*)d_in[4];
  const float* dst_feat   = (const float*)d_in[5];
  const float* memory     = (const float*)d_in[6];
  const float* time_w     = (const float*)d_in[7];
  const float* time_b     = (const float*)d_in[8];
  const float* msg_w1     = (const float*)d_in[9];
  const float* msg_b1     = (const float*)d_in[10];
  const float* msg_w2     = (const float*)d_in[11];
  const float* msg_b2     = (const float*)d_in[12];
  const float* gru_w_ih   = (const float*)d_in[13];
  const float* gru_w_hh   = (const float*)d_in[14];
  const float* gru_b_ih   = (const float*)d_in[15];
  const float* gru_b_hh   = (const float*)d_in[16];
  const float* emb_w1     = (const float*)d_in[17];
  const float* emb_b1     = (const float*)d_in[18];
  const float* emb_w2     = (const float*)d_in[19];
  const float* emb_b2     = (const float*)d_in[20];

  const int E = in_sizes[0];
  const int N = in_sizes[6] / 128;

  char* ws = (char*)d_ws;
  size_t off = 0;
  auto take = [&](size_t bytes) -> void* {
    void* p = ws + off;
    off += (bytes + 255) & ~(size_t)255;
    return p;
  };
  float*    aggr   = (float*)take((size_t)N * 128 * sizeof(float));
  float*    counts = (float*)take((size_t)N * sizeof(float));
  float*    newmem = (float*)take((size_t)N * 128 * sizeof(float));
  _Float16* pW1    = (_Float16*)take((size_t)320 * 128 * 2);
  _Float16* pW2    = (_Float16*)take((size_t)128 * 128 * 2);
  _Float16* pGIH   = (_Float16*)take((size_t)128 * 384 * 2);
  _Float16* pGHH   = (_Float16*)take((size_t)128 * 384 * 2);
  _Float16* pEW1   = (_Float16*)take((size_t)192 * 64 * 2);
  _Float16* pEW2   = (_Float16*)take((size_t)64 * 64 * 2);

  hipMemsetAsync(aggr,   0, (size_t)N * 128 * sizeof(float), stream);
  hipMemsetAsync(counts, 0, (size_t)N * sizeof(float), stream);

  // pack weights into WMMA B-fragment layout (resident in L2 afterwards)
  pack_wmma_b<<<(320 * 128 + 255) / 256, 256, 0, stream>>>(msg_w1,   pW1, 320, 128, 10, 0);
  pack_wmma_b<<<(128 * 128 + 255) / 256, 256, 0, stream>>>(msg_w2,   pW2, 128, 128, 4,  0);
  pack_wmma_b<<<(128 * 384 + 255) / 256, 256, 0, stream>>>(gru_w_ih, pGIH,128, 384, 4,  1);
  pack_wmma_b<<<(128 * 384 + 255) / 256, 256, 0, stream>>>(gru_w_hh, pGHH,128, 384, 4,  1);
  pack_wmma_b<<<(192 * 64  + 255) / 256, 256, 0, stream>>>(emb_w1,   pEW1,192, 64,  6,  0);
  pack_wmma_b<<<(64 * 64   + 255) / 256, 256, 0, stream>>>(emb_w2,   pEW2,64,  64,  2,  0);

  const int eblocks = (E + 15) / 16;
  const int nblocks = (N + 15) / 16;

  tgn_edge_msg<<<eblocks, 256, 0, stream>>>(
      src_nodes, dst_nodes, edge_feat, timestamps, memory, time_w, time_b,
      pW1, msg_b1, pW2, msg_b2, aggr, counts, E);

  tgn_gru<<<nblocks, 256, 0, stream>>>(
      memory, aggr, counts, pGIH, pGHH, gru_b_ih, gru_b_hh, newmem, N);

  tgn_embed<<<eblocks, 256, 0, stream>>>(
      src_nodes, dst_nodes, src_feat, dst_feat, newmem,
      pEW1, emb_b1, pEW2, emb_b2, (float*)d_out, E);
}
